// MultiHeadAttention_33913061769963
// MI455X (gfx1250) — matrix operand order repaired and checked by reading, not compiled, not run on hardware
//
#include <hip/hip_runtime.h>

// ---------------------------------------------------------------------------
// Fused multi-head attention for MI455X (gfx1250, wave32, WMMA).
// B=2, N=2048, D=1024, H=16, DK=64.  ~103 GFLOP, all data resident in L2.
// GEMMs: 128x128 block tiles, LDS double-buffered via
// global_load_async_to_lds_b128 (ASYNCcnt), 8 waves x (32x64) register tiles,
// all MACs on v_wmma_f32_16x16x32_bf16.
// ---------------------------------------------------------------------------

#define BATCH  2
#define SEQ    2048
#define DMODEL 1024
#define HEADS  16
#define DHEAD  64

typedef __attribute__((ext_vector_type(16))) __bf16 v16bf;
typedef __attribute__((ext_vector_type(8)))  float  v8f;

union FragAB {           // 16 bf16 = 32 bytes = 8 VGPRs
    uint4 q[2];
    v16bf v;
};

__device__ __forceinline__ v8f wmma_bf16(const FragAB& a, const FragAB& b, v8f c) {
    return __builtin_amdgcn_wmma_f32_16x16x32_bf16(
        /*neg_a=*/false, a.v, /*neg_b=*/false, b.v,
        /*c_mod=*/(short)0, c, /*reuse_a=*/false, /*reuse_b=*/false);
}

__device__ __forceinline__ unsigned short f32_to_bf16(float f) {
    unsigned u = __float_as_uint(f);
    unsigned r = u + 0x7FFFu + ((u >> 16) & 1u);   // round-to-nearest-even
    return (unsigned short)(r >> 16);
}
__device__ __forceinline__ unsigned pack_bf16(float lo, float hi) {
    return (unsigned)f32_to_bf16(lo) | ((unsigned)f32_to_bf16(hi) << 16);
}
__device__ __forceinline__ uint4 pack8_bf16(const float* s) {
    uint4 o;
    o.x = pack_bf16(s[0], s[1]);
    o.y = pack_bf16(s[2], s[3]);
    o.z = pack_bf16(s[4], s[5]);
    o.w = pack_bf16(s[6], s[7]);
    return o;
}

// Async 16B copy global -> LDS (tracked by ASYNCcnt, no VGPR staging).
__device__ __forceinline__ void async_b128(unsigned lds_off, const void* gptr) {
    asm volatile("global_load_async_to_lds_b128 %0, %1, off"
                 :: "v"(lds_off), "v"((unsigned long long)(uintptr_t)gptr)
                 : "memory");
}

// --------------------------- f32 -> bf16 convert ---------------------------
__global__ __launch_bounds__(256)
void cvt4(const float4* __restrict__ s, uint2* __restrict__ d, int n4) {
    int i = blockIdx.x * blockDim.x + threadIdx.x;
    int stride = gridDim.x * blockDim.x;
    for (; i < n4; i += stride) {
        float4 f = s[i];
        uint2 o;
        o.x = pack_bf16(f.x, f.y);
        o.y = pack_bf16(f.z, f.w);
        d[i] = o;
    }
}

// --------------------- block-tiled GEMM mainloop (shared) ------------------
// C[128x128] = A[128 x 1024] * W[128 x 1024]^T  (both K-contiguous).
// LDS rows padded to 40 halves (80B) -> conflict-free ds_load_b128 fragments.
// acc[mi][ni]: wave tile = 32(M) x 64(N).
__device__ __forceinline__ void gemm_issue(
        const unsigned short* __restrict__ Aptr,
        const unsigned short* __restrict__ Wptr,
        int gm0, int gn0, int k0, int buf,
        unsigned short (*As)[128][40], unsigned short (*Bs)[128][40]) {
    int t = threadIdx.x;          // 0..255
    int r = t >> 1, half = t & 1; // 128 rows x 2 x 16B per tile
    async_b128((unsigned)(uintptr_t)&As[buf][r][half * 16],
               Aptr + (size_t)(gm0 + r) * DMODEL + k0 + half * 16);
    async_b128((unsigned)(uintptr_t)&Bs[buf][r][half * 16],
               Wptr + (size_t)(gn0 + r) * DMODEL + k0 + half * 16);
}

__device__ __forceinline__ void gemm_mainloop(
        const unsigned short* __restrict__ Aptr,
        const unsigned short* __restrict__ Wptr,
        int gm0, int gn0,
        unsigned short (*As)[128][40], unsigned short (*Bs)[128][40],
        v8f acc[2][4]) {
    const int lane = threadIdx.x & 31;
    const int wid  = threadIdx.x >> 5;
    const int lg = lane >> 4, ln = lane & 15;
    const int wm0 = (wid & 3) * 32;   // 4 waves along M
    const int wn0 = (wid >> 2) * 64;  // 2 waves along N
    const int ka = lg * 16, kbch = lg * 16;

    gemm_issue(Aptr, Wptr, gm0, gn0, 0, 0, As, Bs);

    const int STEPS = DMODEL / 32;    // 32
    for (int it = 0; it < STEPS; ++it) {
        int buf = it & 1;
        if (it + 1 < STEPS) {
            gemm_issue(Aptr, Wptr, gm0, gn0, (it + 1) * 32, buf ^ 1, As, Bs);
            asm volatile("s_wait_asynccnt 0x2" ::: "memory"); // current buf landed
        } else {
            asm volatile("s_wait_asynccnt 0x0" ::: "memory");
        }
        __syncthreads();              // whole tile visible to all waves

        FragAB a[2], b[4];
#pragma unroll
        for (int mi = 0; mi < 2; ++mi) {
            const unsigned short* arow = &As[buf][wm0 + mi * 16 + ln][0];
            a[mi].q[0] = *(const uint4*)(arow + ka);
            a[mi].q[1] = *(const uint4*)(arow + ka + 8);
        }
#pragma unroll
        for (int ni = 0; ni < 4; ++ni) {
            const unsigned short* brow = &Bs[buf][wn0 + ni * 16 + ln][0];
            b[ni].q[0] = *(const uint4*)(brow + kbch);
            b[ni].q[1] = *(const uint4*)(brow + kbch + 8);
        }
#pragma unroll
        for (int mi = 0; mi < 2; ++mi)
#pragma unroll
            for (int ni = 0; ni < 4; ++ni)
                acc[mi][ni] = wmma_bf16(a[mi], b[ni], acc[mi][ni]);

        __syncthreads();              // safe to overwrite buf next iteration
    }
}

// ------------------------------ QKV projection -----------------------------
// blockIdx.z selects the matrix: 0=Q, 1=K (bf16 row-major), 2=V (transposed
// store Vt[b][h][dk][n] so the PV WMMA B-fragment is key-contiguous).
__global__ __launch_bounds__(256)
void gemm_qkv_blk(const unsigned short* __restrict__ xb,
                  const unsigned short* __restrict__ wqb,
                  const unsigned short* __restrict__ wkb,
                  const unsigned short* __restrict__ wvb,
                  unsigned short* __restrict__ Qb,
                  unsigned short* __restrict__ Kb,
                  unsigned short* __restrict__ Vt) {
    __shared__ __align__(16) unsigned short As[2][128][40];
    __shared__ __align__(16) unsigned short Bs[2][128][40];

    const int lane = threadIdx.x & 31;
    const int wid  = threadIdx.x >> 5;
    const int lg = lane >> 4, ln = lane & 15;
    const int wm0 = (wid & 3) * 32, wn0 = (wid >> 2) * 64;
    const int gm0 = blockIdx.y * 128, gn0 = blockIdx.x * 128;
    const int mat = blockIdx.z;
    const unsigned short* w = (mat == 0) ? wqb : (mat == 1) ? wkb : wvb;

    v8f acc[2][4] = {};
    gemm_mainloop(xb, w, gm0, gn0, As, Bs, acc);

    __syncthreads();                          // reuse As region as f32 scratch
    float* ftile = (float*)(&As[0][0][0]) + wid * 320;   // 16x20 per wave

#pragma unroll
    for (int mi = 0; mi < 2; ++mi)
#pragma unroll
        for (int ni = 0; ni < 4; ++ni) {
#pragma unroll
            for (int r = 0; r < 8; ++r)
                ftile[(r + 8 * lg) * 20 + ln] = acc[mi][ni][r];
            asm volatile("s_wait_dscnt 0" ::: "memory");
            if (mat < 2) {
                unsigned short* dst = (mat == 0) ? Qb : Kb;
                int row = ln, ch = lg * 8;
                float v[8];
#pragma unroll
                for (int j = 0; j < 8; ++j) v[j] = ftile[row * 20 + ch + j];
                *(uint4*)(dst + (size_t)(gm0 + wm0 + mi * 16 + row) * DMODEL +
                          gn0 + wn0 + ni * 16 + ch) = pack8_bf16(v);
            } else {
                int ocol = ln, nh = lg * 8;
                float v[8];
#pragma unroll
                for (int j = 0; j < 8; ++j) v[j] = ftile[(nh + j) * 20 + ocol];
                int og = gn0 + wn0 + ni * 16 + ocol;
                int h = og >> 6, dk = og & (DHEAD - 1);
                int ng = gm0 + wm0 + mi * 16 + nh;
                int bb = ng >> 11, nn = ng & (SEQ - 1);
                *(uint4*)(Vt + ((size_t)((bb * HEADS + h) * DHEAD + dk)) * SEQ + nn)
                    = pack8_bf16(v);
            }
        }
}

// ------------------------------ output projection --------------------------
__global__ __launch_bounds__(256)
void gemm_out_blk(const unsigned short* __restrict__ ctxb,
                  const unsigned short* __restrict__ wob,
                  const float* __restrict__ bout,
                  float* __restrict__ out) {
    __shared__ __align__(16) unsigned short As[2][128][40];
    __shared__ __align__(16) unsigned short Bs[2][128][40];

    const int lane = threadIdx.x & 31;
    const int wid  = threadIdx.x >> 5;
    const int lg = lane >> 4, ln = lane & 15;
    const int wm0 = (wid & 3) * 32, wn0 = (wid >> 2) * 64;
    const int gm0 = blockIdx.y * 128, gn0 = blockIdx.x * 128;

    v8f acc[2][4] = {};
    gemm_mainloop(ctxb, wob, gm0, gn0, As, Bs, acc);

    __syncthreads();
    float* ftile = (float*)(&As[0][0][0]) + wid * 320;

#pragma unroll
    for (int mi = 0; mi < 2; ++mi)
#pragma unroll
        for (int ni = 0; ni < 4; ++ni) {
#pragma unroll
            for (int r = 0; r < 8; ++r)
                ftile[(r + 8 * lg) * 20 + ln] = acc[mi][ni][r];
            asm volatile("s_wait_dscnt 0" ::: "memory");
            int row = ln, ch = lg * 8;
            int oc = gn0 + wn0 + ni * 16 + ch;
            float* dst = out + (size_t)(gm0 + wm0 + mi * 16 + row) * DMODEL + oc;
#pragma unroll
            for (int j = 0; j < 8; ++j)
                dst[j] = ftile[row * 20 + ch + j] + bout[oc + j];
        }
}

// ------------------------------ flash attention ----------------------------
// One wave = one (b, h, 16-query tile).  Online softmax over causal keys.
__global__ __launch_bounds__(128)
void attn(const unsigned short* __restrict__ Qb,
          const unsigned short* __restrict__ Kb,
          const unsigned short* __restrict__ Vt,
          const int* __restrict__ amask,
          unsigned short* __restrict__ ctxb) {
    __shared__ __align__(16) unsigned short Plds[4][16][40];
    __shared__ __align__(16) float Clds[4][16][68];

    const int lane = threadIdx.x & 31;
    const int wid  = threadIdx.x >> 5;
    const int lg = lane >> 4, ln = lane & 15;

    int t = blockIdx.x * 4 + wid;                       // 0..4095
    const int NTQ = SEQ / 16;                           // 128
    int bb = t / (HEADS * NTQ);
    int rem = t % (HEADS * NTQ);
    int h = rem / NTQ, i = rem % NTQ;
    int m0 = i * 16;

    // Q A-fragments for d=0..31 and d=32..63
    const unsigned short* qrow =
        Qb + (size_t)(bb * SEQ + m0 + ln) * DMODEL + h * DHEAD;
    FragAB aq0, aq1;
    aq0.q[0] = *(const uint4*)(qrow + lg * 16);
    aq0.q[1] = *(const uint4*)(qrow + lg * 16 + 8);
    aq1.q[0] = *(const uint4*)(qrow + 32 + lg * 16);
    aq1.q[1] = *(const uint4*)(qrow + 32 + lg * 16 + 8);

    const unsigned short* kbase = Kb + (size_t)(bb * SEQ) * DMODEL + h * DHEAD;
    const float NEG = -__builtin_inff();

    v8f acc[4] = {};            // 16 rows x 64 cols context accumulator
    float mrun[8], lrun[8];
#pragma unroll
    for (int r = 0; r < 8; ++r) { mrun[r] = NEG; lrun[r] = 0.0f; }

    for (int kb = 0; kb < m0 + 16; kb += 32) {
        bool have2 = kb < m0;   // second 16-key tile needed (else fully causal-masked)

        // ---- S = Q @ K^T for 32 keys (two 16x16 tiles) ----
        const unsigned short* krow = kbase + (size_t)(kb + ln) * DMODEL;
        FragAB bk0, bk1;
        bk0.q[0] = *(const uint4*)(krow + lg * 16);
        bk0.q[1] = *(const uint4*)(krow + lg * 16 + 8);
        bk1.q[0] = *(const uint4*)(krow + 32 + lg * 16);
        bk1.q[1] = *(const uint4*)(krow + 32 + lg * 16 + 8);
        v8f s0 = {};
        s0 = wmma_bf16(aq0, bk0, s0);
        s0 = wmma_bf16(aq1, bk1, s0);
        v8f s1 = {};
        if (have2) {
            const unsigned short* krow2 = kbase + (size_t)(kb + 16 + ln) * DMODEL;
            FragAB ck0, ck1;
            ck0.q[0] = *(const uint4*)(krow2 + lg * 16);
            ck0.q[1] = *(const uint4*)(krow2 + lg * 16 + 8);
            ck1.q[0] = *(const uint4*)(krow2 + 32 + lg * 16);
            ck1.q[1] = *(const uint4*)(krow2 + 32 + lg * 16 + 8);
            s1 = wmma_bf16(aq0, ck0, s1);
            s1 = wmma_bf16(aq1, ck1, s1);
        }

        int pm0 = amask[bb * SEQ + kb + ln];
        int pm1 = have2 ? amask[bb * SEQ + kb + 16 + ln] : 0;
        int key0 = kb + ln;

        // ---- mask, scale(0.125 after mask), online softmax per row ----
        float p0[8], p1[8];
#pragma unroll
        for (int r = 0; r < 8; ++r) {
            int mg = m0 + r + 8 * lg;
            float v0 = (pm0 && key0 <= mg)            ? s0[r] * 0.125f : NEG;
            float v1 = (pm1 && (key0 + 16) <= mg)     ? s1[r] * 0.125f : NEG;
            float mx = fmaxf(v0, v1);
            mx = fmaxf(mx, __shfl_xor(mx, 1, 32));
            mx = fmaxf(mx, __shfl_xor(mx, 2, 32));
            mx = fmaxf(mx, __shfl_xor(mx, 4, 32));
            mx = fmaxf(mx, __shfl_xor(mx, 8, 32));
            float newm = fmaxf(mrun[r], mx);
            float al = __expf(mrun[r] - newm);        // 0 on first block
            p0[r] = __expf(v0 - newm);
            p1[r] = __expf(v1 - newm);
            float rs = p0[r] + p1[r];
            rs += __shfl_xor(rs, 1, 32);
            rs += __shfl_xor(rs, 2, 32);
            rs += __shfl_xor(rs, 4, 32);
            rs += __shfl_xor(rs, 8, 32);
            lrun[r] = lrun[r] * al + rs;
            mrun[r] = newm;
            acc[0][r] *= al; acc[1][r] *= al; acc[2][r] *= al; acc[3][r] *= al;
            int Mr = r + 8 * lg;
            Plds[wid][Mr][ln]      = f32_to_bf16(p0[r]);
            Plds[wid][Mr][16 + ln] = f32_to_bf16(p1[r]);
        }
        asm volatile("s_wait_dscnt 0" ::: "memory");

        // re-layout P (D-frag) -> A-frag via LDS
        FragAB ap;
        const unsigned short* prow = &Plds[wid][ln][0];
    ap.q[0] = *(const uint4*)(prow + lg * 16);
    ap.q[1] = *(const uint4*)(prow + lg * 16 + 8);

        // ---- ctx += P @ V ----  (V stored [B,H,DK,N]: key-contiguous)
        int kvb = kb + lg * 16;
        if (!have2 && lg) kvb = kb;      // clamp OOB block; its P columns are 0
#pragma unroll
        for (int tt = 0; tt < 4; ++tt) {
            const unsigned short* vrow =
                Vt + ((size_t)((bb * HEADS + h) * DHEAD + tt * 16 + ln)) * SEQ + kvb;
            FragAB bv;
            bv.q[0] = *(const uint4*)(vrow);
            bv.q[1] = *(const uint4*)(vrow + 8);
            acc[tt] = wmma_bf16(ap, bv, acc[tt]);
        }
    }

    // ---- normalize, write ctx bf16 [B*N, D] via LDS re-layout ----
    float inv[8];
#pragma unroll
    for (int r = 0; r < 8; ++r) inv[r] = 1.0f / lrun[r];
#pragma unroll
    for (int tt = 0; tt < 4; ++tt)
#pragma unroll
        for (int r = 0; r < 8; ++r)
            Clds[wid][r + 8 * lg][tt * 16 + ln] = acc[tt][r] * inv[r];
    asm volatile("s_wait_dscnt 0" ::: "memory");

    int row = ln, half = lg;
    unsigned short* crow =
        ctxb + (size_t)(bb * SEQ + m0 + row) * DMODEL + h * DHEAD + half * 32;
    const float* src = &Clds[wid][row][half * 32];
#pragma unroll
    for (int j = 0; j < 4; ++j)
        *(uint4*)(crow + j * 8) = pack8_bf16(src + j * 8);
}

// ------------------------------ launcher -----------------------------------
extern "C" void kernel_launch(void* const* d_in, const int* in_sizes, int n_in,
                              void* d_out, int out_size, void* d_ws, size_t ws_size,
                              hipStream_t stream) {
    const float* x     = (const float*)d_in[0];
    const int*   amask = (const int*)  d_in[1];
    const float* W_Q   = (const float*)d_in[2];
    const float* W_K   = (const float*)d_in[3];
    const float* W_V   = (const float*)d_in[4];
    const float* W_out = (const float*)d_in[5];
    const float* b_out = (const float*)d_in[6];
    float* out = (float*)d_out;

    const size_t XN = (size_t)BATCH * SEQ * DMODEL;   // 4M elems
    const size_t WN = (size_t)DMODEL * DMODEL;        // 1M elems

    // workspace carve-up (bf16 halves): ~48 MB total
    unsigned short* xb  = (unsigned short*)d_ws;
    unsigned short* wqb = xb  + XN;
    unsigned short* wkb = wqb + WN;
    unsigned short* wvb = wkb + WN;
    unsigned short* wob = wvb + WN;
    unsigned short* Qb  = wob + WN;
    unsigned short* Kb  = Qb  + XN;
    unsigned short* Vt  = Kb  + XN;
    unsigned short* ctx = Vt  + XN;

    cvt4<<<1024, 256, 0, stream>>>((const float4*)x,     (uint2*)xb,  (int)(XN / 4));
    cvt4<<<512,  256, 0, stream>>>((const float4*)W_Q,   (uint2*)wqb, (int)(WN / 4));
    cvt4<<<512,  256, 0, stream>>>((const float4*)W_K,   (uint2*)wkb, (int)(WN / 4));
    cvt4<<<512,  256, 0, stream>>>((const float4*)W_V,   (uint2*)wvb, (int)(WN / 4));
    cvt4<<<512,  256, 0, stream>>>((const float4*)W_out, (uint2*)wob, (int)(WN / 4));

    // QKV: grid (N/128, M/128, 3 matrices), 128x128 block tiles
    gemm_qkv_blk<<<dim3(8, 32, 3), 256, 0, stream>>>(xb, wqb, wkb, wvb, Qb, Kb, Vt);

    // 2 * 16 * 128 (b,h,qtile) waves / 4 per block
    attn<<<1024, 128, 0, stream>>>(Qb, Kb, Vt, amask, ctx);

    // out projection: grid (8, 32)
    gemm_out_blk<<<dim3(8, 32), 256, 0, stream>>>(ctx, wob, b_out, out);
}
